// DLREM_16501264351579
// MI455X (gfx1250) — matrix operand order repaired and checked by reading, not compiled, not run on hardware
//
#include <hip/hip_runtime.h>

// ---------------------------------------------------------------------------
// CDNA5 (gfx1250) wave32 WMMA building blocks, fragment-ordered tiles
// ---------------------------------------------------------------------------
typedef __attribute__((ext_vector_type(16))) _Float16 v16h;
typedef __attribute__((ext_vector_type(4)))  _Float16 v4h;
typedef __attribute__((ext_vector_type(8)))  float    v8f;
typedef __attribute__((ext_vector_type(4)))  float    v4f;

__device__ __forceinline__ v8f wmma_f16(v16h a, v16h b, v8f c) {
  return __builtin_amdgcn_wmma_f32_16x16x32_f16(
      false, a, false, b, (short)0, c, false, false);
}

// Fragment-ordered tile: [kslice][lane][16] halfs. One lane's fragment is 16
// contiguous halfs (32B) -> loads lower to 2x ds_load_b128 / global_load_b128.
// A-slot of element (m, k) per ISA A-layout (lanes0-15: K{0..7,16..23}, etc.)
__device__ __forceinline__ int a_slot(int m, int k) {
  int kk = k >> 5, kr = k & 31;
  int hi = (kr >> 3) & 1;
  int ln = m + (hi << 4);
  int krr = kr - (hi << 3);
  int e = (krr >= 16) ? (krr - 8) : krr;
  return (kk * 32 + ln) * 16 + e;          // 4-aligned k -> 4 contiguous slots
}

__device__ __forceinline__ v16h frag_ld(const _Float16* p) {
  return *(const v16h*)p;                  // 32B contiguous
}

__device__ __forceinline__ float actf(float y, int act) {
  if (act == 1) return fmaxf(y, 0.f);
  if (act == 2) return (y > 0.f) ? y : 0.2f * y;
  return y;
}

// C/D layout: lanes0-15: N=lane, M=v ; lanes16-31: N=lane-16, M=v+8
__device__ __forceinline__ void store_c(float* out, int ldo, v8f c,
                                        const float* bias, int n0, int act) {
  int lane = threadIdx.x & 31;
  int n = lane & 15, mb = (lane >> 4) << 3;
  float bv = bias[n0 + n];
#pragma unroll
  for (int v = 0; v < 8; ++v)
    out[(size_t)(mb + v) * ldo + n0 + n] = actf(c[v] + bv, act);
}

// f32 W[N][K] (row-major, torch Linear) -> f16 fragment-ordered B tiles:
// out[((kk*nct+ct)*32 + lane)*16 + e], lanes0-15: K{0..7,16..23}, 16-31: +8
__global__ void k_wprep(const float* __restrict__ W, _Float16* __restrict__ out,
                        int K, int N) {
  int i = blockIdx.x * blockDim.x + threadIdx.x;
  if (i >= K * N) return;
  int e = i & 15, lane = (i >> 4) & 31, tile = i >> 9;
  int nct = N >> 4;
  int kk = tile / nct, ct = tile - kk * nct;
  int n = ct * 16 + (lane & 15);
  int k = kk * 32 + ((lane >> 4) << 3) + e + ((e >> 3) << 3);
  out[i] = (_Float16)W[(size_t)n * K + k];
}

// ---------------------------------------------------------------------------
// Fused gather -> concat(128) -> Lin(128,64)+Leaky -> Lin(64,64)  (+ mask)
// 1 wave = 16 rows; 256 threads = 8 waves = 128 rows / block.
// ---------------------------------------------------------------------------
__global__ __launch_bounds__(256) void k_gmlp(
    const float* __restrict__ embA, const float* __restrict__ embR,
    const int* __restrict__ idx,
    const _Float16* __restrict__ W1f, const float* __restrict__ b1,
    const _Float16* __restrict__ W2f, const float* __restrict__ b2,
    float* __restrict__ out, float* __restrict__ mask, int nrows)
{
  __shared__ _Float16 sA[8][16 * 128];
  int lane = threadIdx.x & 31, wave = threadIdx.x >> 5;
  int r0 = (blockIdx.x * 8 + wave) * 16;
  if (r0 >= nrows) return;
  _Float16* A = sA[wave];

  // stage: float4 gather -> v4h fragment-order scatter (ds_store_b64)
  for (int i = lane; i < 16 * 32; i += 32) {
    int r = i >> 5, q = i & 31, c4 = q * 4;
    int rr = r0 + r; if (rr >= nrows) rr = nrows - 1;
    int i0 = idx[rr * 2 + 0], i1 = idx[rr * 2 + 1];
    v4f f = (c4 < 64) ? *(const v4f*)(embA + (size_t)i0 * 64 + c4)
                      : *(const v4f*)(embR + (size_t)i1 * 64 + (c4 - 64));
    v4h pk = { (_Float16)f.x, (_Float16)f.y, (_Float16)f.z, (_Float16)f.w };
    *(v4h*)(A + a_slot(r, c4)) = pk;
    if (q == 0) mask[rr] = (i0 > 0) ? 1.f : 0.f;
  }

  v16h a0[4];
#pragma unroll
  for (int kk = 0; kk < 4; ++kk) a0[kk] = frag_ld(A + (kk * 32 + lane) * 16);

  v8f h[4];
#pragma unroll
  for (int ct = 0; ct < 4; ++ct) {
    v8f acc = {};
#pragma unroll
    for (int kk = 0; kk < 4; ++kk)
      acc = wmma_f16(a0[kk], frag_ld(W1f + ((size_t)(kk * 4 + ct) * 32 + lane) * 16), acc);
    h[ct] = acc;
  }
  // bias+leaky, restage hidden as fragment-order f16 (wave-local, DS in-order)
  {
    int n = lane & 15, mb = (lane >> 4) << 3;
#pragma unroll
    for (int ct = 0; ct < 4; ++ct) {
      float bv = b1[ct * 16 + n];
#pragma unroll
      for (int v = 0; v < 8; ++v)
        A[a_slot(mb + v, ct * 16 + n)] = (_Float16)actf(h[ct][v] + bv, 2);
    }
  }
  v16h a1[2];
#pragma unroll
  for (int kk = 0; kk < 2; ++kk) a1[kk] = frag_ld(A + (kk * 32 + lane) * 16);
#pragma unroll
  for (int ct = 0; ct < 4; ++ct) {
    v8f acc = {};
#pragma unroll
    for (int kk = 0; kk < 2; ++kk)
      acc = wmma_f16(a1[kk], frag_ld(W2f + ((size_t)(kk * 4 + ct) * 32 + lane) * 16), acc);
    store_c(out + (size_t)r0 * 64, 64, acc, b2, ct * 16, 0);
  }
}

// ---------------------------------------------------------------------------
// Generic Linear (templated on DIN for static unroll): y = act(x@W^T + b)
// optional dual 64-wide concat input. 128 threads = 4 waves = 64 rows.
// ---------------------------------------------------------------------------
template <int DIN>
__global__ __launch_bounds__(128) void k_lin(
    const float* __restrict__ xa, const float* __restrict__ xb, int dout,
    const _Float16* __restrict__ Wf, const float* __restrict__ bias,
    float* __restrict__ out, int nrows, int act)
{
  __shared__ _Float16 sA[4][16 * DIN];
  int lane = threadIdx.x & 31, wave = threadIdx.x >> 5;
  int r0 = (blockIdx.x * 4 + wave) * 16;
  if (r0 >= nrows) return;
  _Float16* A = sA[wave];
  constexpr int NK = DIN / 32;

  for (int i = lane; i < 16 * (DIN / 4); i += 32) {
    int r = i / (DIN / 4), c4 = (i % (DIN / 4)) * 4;
    size_t row = (size_t)(r0 + r);
    v4f f = xb ? ((c4 < 64) ? *(const v4f*)(xa + row * 64 + c4)
                            : *(const v4f*)(xb + row * 64 + (c4 - 64)))
               : *(const v4f*)(xa + row * DIN + c4);
    v4h pk = { (_Float16)f.x, (_Float16)f.y, (_Float16)f.z, (_Float16)f.w };
    *(v4h*)(A + a_slot(r, c4)) = pk;
  }

  v16h a[NK];
#pragma unroll
  for (int kk = 0; kk < NK; ++kk) a[kk] = frag_ld(A + (kk * 32 + lane) * 16);

  int nct = dout >> 4;
  for (int ct = 0; ct < nct; ++ct) {
    v8f acc = {};
#pragma unroll
    for (int kk = 0; kk < NK; ++kk)
      acc = wmma_f16(a[kk], frag_ld(Wf + ((size_t)(kk * nct + ct) * 32 + lane) * 16), acc);
    store_c(out + (size_t)r0 * dout, dout, acc, bias, ct * 16, act);
  }
}

// ---------------------------------------------------------------------------
// Full 4-layer GRU on-chip. One wave owns 16 rows for all layers/steps.
// h kept in VGPRs; fragment-order LDS only to turn h into A-fragments.
// Inter-layer sequence stored fragment-ordered f16 -> layers>0 load X
// fragments directly from global (global_load_b128, same-wave ordered).
// ---------------------------------------------------------------------------
__global__ __launch_bounds__(256) void k_gru(
    const float* __restrict__ x, long long sb, long long sj, long long st,
    const float* __restrict__ h0,                 // [4,256,64] f32
    const _Float16* __restrict__ Wihf,            // frag-ordered [4][tiles]
    const _Float16* __restrict__ Whhf,
    const float* __restrict__ bih, const float* __restrict__ bhh, // [4][192]
    _Float16* __restrict__ seqf,                  // [nrows/16][T][2][512] halfs
    float* __restrict__ outlast, int nrows, int T, int outL)
{
  __shared__ _Float16 sX[8][2 * 512];
  __shared__ _Float16 sH[8][2 * 512];
  int lane = threadIdx.x & 31, wave = threadIdx.x >> 5;
  int tile = blockIdx.x * 8 + wave;
  int r0 = tile * 16;
  if (r0 >= nrows) return;
  _Float16* X = sX[wave];
  _Float16* H = sH[wave];
  int n = lane & 15, mb = (lane >> 4) << 3;

#pragma unroll 1
  for (int l = 0; l < 4; ++l) {
    const _Float16* wi = Wihf + (size_t)l * 64 * 192;
    const _Float16* wh = Whhf + (size_t)l * 64 * 192;
    const float* bi = bih + l * 192;
    const float* bh = bhh + l * 192;
    float bir[4], biz[4], bin[4], bhr[4], bhz[4], bhn[4];
    v8f hprev[4];
#pragma unroll
    for (int c = 0; c < 4; ++c) {
      bir[c] = bi[c * 16 + n];  biz[c] = bi[64 + c * 16 + n];  bin[c] = bi[128 + c * 16 + n];
      bhr[c] = bh[c * 16 + n];  bhz[c] = bh[64 + c * 16 + n];  bhn[c] = bh[128 + c * 16 + n];
      v8f hv;
#pragma unroll
      for (int v = 0; v < 8; ++v) {
        int b = (r0 + mb + v) & 255;
        float f = h0[((size_t)l * 256 + b) * 64 + c * 16 + n];
        hv[v] = f;
        H[a_slot(mb + v, c * 16 + n)] = (_Float16)f;
      }
      hprev[c] = hv;
    }

#pragma unroll 1
    for (int t = 0; t < T; ++t) {
      v16h ax0, ax1;
      if (l == 0) {
        for (int i = lane; i < 16 * 16; i += 32) {
          int r = i >> 4, c4 = (i & 15) * 4;
          long long row = r0 + r, b = row & 255, j = row >> 8;
          v4f f = *(const v4f*)(x + b * sb + j * sj + (long long)t * st + c4);
          v4h pk = { (_Float16)f.x, (_Float16)f.y, (_Float16)f.z, (_Float16)f.w };
          *(v4h*)(X + a_slot(r, c4)) = pk;
        }
        ax0 = frag_ld(X + lane * 16);
        ax1 = frag_ld(X + (32 + lane) * 16);
      } else {
        const _Float16* sp = seqf + (((size_t)tile * T + t) * 2) * 512;
        ax0 = frag_ld(sp + lane * 16);
        ax1 = frag_ld(sp + 512 + lane * 16);
      }
      v16h ah0 = frag_ld(H + lane * 16);
      v16h ah1 = frag_ld(H + (32 + lane) * 16);

      _Float16* sp = seqf + (((size_t)tile * T + t) * 2) * 512;
#pragma unroll
      for (int c = 0; c < 4; ++c) {
        v8f ir = {}, iz = {}, in_ = {}, hr = {}, hz = {}, hn = {};
        ir  = wmma_f16(ax0, frag_ld(wi + ((size_t)(0 * 12 + 0 * 4 + c) * 32 + lane) * 16), ir);
        ir  = wmma_f16(ax1, frag_ld(wi + ((size_t)(1 * 12 + 0 * 4 + c) * 32 + lane) * 16), ir);
        iz  = wmma_f16(ax0, frag_ld(wi + ((size_t)(0 * 12 + 1 * 4 + c) * 32 + lane) * 16), iz);
        iz  = wmma_f16(ax1, frag_ld(wi + ((size_t)(1 * 12 + 1 * 4 + c) * 32 + lane) * 16), iz);
        in_ = wmma_f16(ax0, frag_ld(wi + ((size_t)(0 * 12 + 2 * 4 + c) * 32 + lane) * 16), in_);
        in_ = wmma_f16(ax1, frag_ld(wi + ((size_t)(1 * 12 + 2 * 4 + c) * 32 + lane) * 16), in_);
        hr  = wmma_f16(ah0, frag_ld(wh + ((size_t)(0 * 12 + 0 * 4 + c) * 32 + lane) * 16), hr);
        hr  = wmma_f16(ah1, frag_ld(wh + ((size_t)(1 * 12 + 0 * 4 + c) * 32 + lane) * 16), hr);
        hz  = wmma_f16(ah0, frag_ld(wh + ((size_t)(0 * 12 + 1 * 4 + c) * 32 + lane) * 16), hz);
        hz  = wmma_f16(ah1, frag_ld(wh + ((size_t)(1 * 12 + 1 * 4 + c) * 32 + lane) * 16), hz);
        hn  = wmma_f16(ah0, frag_ld(wh + ((size_t)(0 * 12 + 2 * 4 + c) * 32 + lane) * 16), hn);
        hn  = wmma_f16(ah1, frag_ld(wh + ((size_t)(1 * 12 + 2 * 4 + c) * 32 + lane) * 16), hn);
        v8f hv;
#pragma unroll
        for (int v = 0; v < 8; ++v) {
          float rg = 1.f / (1.f + __expf(-(ir[v] + bir[c] + hr[v] + bhr[c])));
          float zg = 1.f / (1.f + __expf(-(iz[v] + biz[c] + hz[v] + bhz[c])));
          float ng = tanhf(in_[v] + bin[c] + rg * (hn[v] + bhn[c]));
          hv[v] = (1.f - zg) * ng + zg * hprev[c][v];
        }
        hprev[c] = hv;
#pragma unroll
        for (int v = 0; v < 8; ++v) {
          int sl = a_slot(mb + v, c * 16 + n);
          _Float16 h16 = (_Float16)hv[v];
          H[sl] = h16;
          sp[sl] = h16;
        }
      }
    }
    // top layer: emit last hidden state (f32)
    if (l == 3) {
#pragma unroll
      for (int c = 0; c < 4; ++c)
#pragma unroll
        for (int v = 0; v < 8; ++v) {
          long long row = r0 + mb + v;
          long long orow = outL ? ((row & 255) * (long long)outL + (row >> 8)) : row;
          outlast[orow * 64 + c * 16 + n] = hprev[c][v];
        }
    }
  }
}

// ---------------------------------------------------------------------------
// Small VALU kernels
// ---------------------------------------------------------------------------
__global__ void k_dot(const float* __restrict__ x, int din,
                      const float* __restrict__ w, const float* __restrict__ b,
                      float* __restrict__ out, int nrows)
{
  int wid = blockIdx.x * (blockDim.x >> 5) + (threadIdx.x >> 5);
  if (wid >= nrows) return;
  int lane = threadIdx.x & 31;
  float s = 0.f;
  for (int k = lane; k < din; k += 32) s += x[(size_t)wid * din + k] * w[k];
  for (int o = 16; o; o >>= 1) s += __shfl_xor(s, o, 32);
  if (lane == 0) out[wid] = s + b[0];
}

__global__ void k_attwsum(const float* __restrict__ s, const float* __restrict__ mask,
                          const float* __restrict__ x, float* __restrict__ out,
                          int nouter, int linner)
{
  int wid = blockIdx.x * (blockDim.x >> 5) + (threadIdx.x >> 5);
  if (wid >= nouter) return;
  int lane = threadIdx.x & 31;
  float w = 0.f;
  if (lane < linner) {
    size_t i = (size_t)wid * linner + lane;
    w = __expf(s[i]) * mask[i];
  }
  float tot = w;
  for (int o = 16; o; o >>= 1) tot += __shfl_xor(tot, o, 32);
  float alpha = w / (tot + 1e-10f);
  float a0 = 0.f, a1 = 0.f;
  for (int l = 0; l < linner; ++l) {
    float al = __shfl(alpha, l, 32);
    const float* xr = x + ((size_t)wid * linner + l) * 64;
    a0 += al * xr[lane];
    a1 += al * xr[32 + lane];
  }
  out[(size_t)wid * 64 + lane] = a0;
  out[(size_t)wid * 64 + 32 + lane] = a1;
}

__global__ void k_mul(const float* a, const float* b, float* o, int n) {
  int i = blockIdx.x * blockDim.x + threadIdx.x;
  if (i < n) o[i] = a[i] * b[i];
}
__global__ void k_mulb(const float* a, const float* bb, float* o, int nrows, int L) {
  int i = blockIdx.x * blockDim.x + threadIdx.x;
  if (i >= nrows * 64) return;
  int r = i >> 6, d = i & 63;
  o[i] = a[i] * bb[(size_t)(r / L) * 64 + d];
}
__global__ void k_pmask(const float* emb, const int* gid, const float* mask,
                        float* o, int nrows, int L) {
  int i = blockIdx.x * blockDim.x + threadIdx.x;
  if (i >= nrows * 64) return;
  int r = i >> 6, d = i & 63;
  o[i] = mask[r] * emb[(size_t)gid[r / L] * 64 + d];
}
__global__ void k_gatherE(const float* emb, const int* gid, float* o, int nrows) {
  int i = blockIdx.x * blockDim.x + threadIdx.x;
  if (i >= nrows * 64) return;
  int r = i >> 6, d = i & 63;
  o[i] = emb[(size_t)gid[r] * 64 + d];
}
__global__ void k_maskint(const int* gid, float* o, int n) {
  int i = blockIdx.x * blockDim.x + threadIdx.x;
  if (i < n) o[i] = (gid[i] > 0) ? 1.f : 0.f;
}

// ---------------------------------------------------------------------------
// Host side
// ---------------------------------------------------------------------------
struct TowerP {
  const float *w1w,*w1b,*w2w,*w2b,*w3w,*w3b;
  const float *g1w,*g1b,*g2w,*g2b;
  const float *a1w,*a1b,*a1vw,*a1vb;
  const float *a2w,*a2b,*a2vw,*a2vb;
  const float *a3w,*a3b,*a3vw,*a3vb;
  const float *ag1w,*ag1b,*ag2w,*ag2b,*ag3w,*ag3b;
  const float *gwih[4],*gwhh[4],*gbih[4],*gbhh[4];
  const float *h0;
  const float *c1w,*c1b,*c2w,*c2b;
};

static TowerP parse_tower(void* const* d_in, int& c) {
  TowerP t;
  auto F = [&]() { return (const float*)d_in[c++]; };
  t.w1w=F(); t.w1b=F(); t.w2w=F(); t.w2b=F(); t.w3w=F(); t.w3b=F();
  t.g1w=F(); t.g1b=F(); t.g2w=F(); t.g2b=F();
  t.a1w=F(); t.a1b=F(); t.a1vw=F(); t.a1vb=F();
  t.a2w=F(); t.a2b=F(); t.a2vw=F(); t.a2vb=F();
  t.a3w=F(); t.a3b=F(); t.a3vw=F(); t.a3vb=F();
  t.ag1w=F(); t.ag1b=F(); t.ag2w=F(); t.ag2b=F(); t.ag3w=F(); t.ag3b=F();
  for (int l = 0; l < 4; ++l) { t.gwih[l]=F(); t.gwhh[l]=F(); t.gbih[l]=F(); t.gbhh[l]=F(); }
  t.h0 = F();
  t.c1w=F(); t.c1b=F(); t.c2w=F(); t.c2b=F();
  return t;
}

struct Bufs {
  float *xL1,*mL1,*tA1,*tB1,*s1,*sum1,*hL1,*hS1,*hI1;
  float *xL2,*mL2,*tA2,*tB2,*s2,*sum2,*hL2,*hS2,*hI2;
  float *mSU,*s3,*sum3,*hN,*c1;
  _Float16 *seqf;                 // aliases tB2 (disjoint lifetime)
  _Float16 *w1f,*w2f,*w3f,*g1f,*g2f,*a1f,*a2f,*a3f,*ag1f,*ag2f,*ag3f,*c1f,*c2f;
  _Float16 *wihf,*whhf;
  float *bihp,*bhhp;
};

static void prep_w(const float* W, _Float16* out, int K, int N, hipStream_t s) {
  int tot = K * N;
  k_wprep<<<(tot + 255) / 256, 256, 0, s>>>(W, out, K, N);
}

static void run_tower(const TowerP& T, bool is_user,
    const float* emb_q, const float* emb_p, const float* rate_emb,
    const int* ids, const int* l1_pad, const int* nb_pad, const int* l2_pad,
    const Bufs& B, float* towerOut, hipStream_t s)
{
  const int BN = 256, L = 30, BL = BN * L, BLL = BL * L;

  // weights -> fragment-ordered f16 (once per tower)
  prep_w(T.w1w, B.w1f, 64, 64, s);   prep_w(T.w2w, B.w2f, 64, 64, s);
  prep_w(T.w3w, B.w3f, 64, 64, s);
  prep_w(T.g1w, B.g1f, 128, 64, s);  prep_w(T.g2w, B.g2f, 64, 64, s);
  prep_w(T.a1w, B.a1f, 128, 64, s);  prep_w(T.a2w, B.a2f, 128, 64, s);
  prep_w(T.a3w, B.a3f, 128, 64, s);
  prep_w(T.ag1w, B.ag1f, 64, 64, s); prep_w(T.ag2w, B.ag2f, 64, 64, s);
  prep_w(T.ag3w, B.ag3f, 64, 64, s);
  prep_w(T.c1w, B.c1f, 128, 128, s); prep_w(T.c2w, B.c2f, 128, 64, s);
  for (int l = 0; l < 4; ++l) {
    prep_w(T.gwih[l], B.wihf + (size_t)l * 64 * 192, 64, 192, s);
    prep_w(T.gwhh[l], B.whhf + (size_t)l * 64 * 192, 64, 192, s);
    hipMemcpyAsync(B.bihp + l * 192, T.gbih[l], 192 * sizeof(float), hipMemcpyDeviceToDevice, s);
    hipMemcpyAsync(B.bhhp + l * 192, T.gbhh[l], 192 * sizeof(float), hipMemcpyDeviceToDevice, s);
  }

  // ---------------- level 1 ----------------
  k_gmlp<<<BL / 128, 256, 0, s>>>(emb_q, rate_emb, l1_pad, B.g1f, T.g1b, B.g2f, T.g2b,
                                  B.xL1, B.mL1, BL);
  k_lin<64><<<BL / 64, 128, 0, s>>>(B.xL1, nullptr, 64, B.w1f, T.w1b, B.tA1, BL, 0);
  k_pmask<<<(BL * 64 + 255) / 256, 256, 0, s>>>(emb_p, ids, B.mL1, B.tB1, BL, L);
  k_lin<64><<<BL / 64, 128, 0, s>>>(B.tB1, nullptr, 64, B.w1f, T.w1b, B.tB1, BL, 0);
  k_lin<128><<<BL / 64, 128, 0, s>>>(B.tA1, B.tB1, 64, B.a1f, T.a1b, B.tB1, BL, 2);
  k_dot<<<(BL + 7) / 8, 256, 0, s>>>(B.tB1, 64, T.a1vw, T.a1vb, B.s1, BL);
  k_attwsum<<<(BN + 7) / 8, 256, 0, s>>>(B.s1, B.mL1, is_user ? B.xL1 : B.tA1, B.sum1, BN, L);
  k_lin<64><<<BN / 64, 128, 0, s>>>(B.sum1, nullptr, 64, B.ag1f, T.ag1b, B.hL1, BN, 1);
  k_gru<<<BN / 128, 256, 0, s>>>(B.xL1, 30LL * 64, 0LL, 64LL, T.h0,
                                 B.wihf, B.whhf, B.bihp, B.bhhp, B.seqf, B.hS1, BN, 30, 0);
  k_mul<<<(BN * 64 + 255) / 256, 256, 0, s>>>(B.hL1, B.hS1, B.hI1, BN * 64);

  // ---------------- level 2 ----------------
  k_gmlp<<<BLL / 128, 256, 0, s>>>(emb_q, rate_emb, l2_pad, B.g1f, T.g1b, B.g2f, T.g2b,
                                   B.xL2, B.mL2, BLL);
  k_lin<64><<<BLL / 64, 128, 0, s>>>(B.xL2, nullptr, 64, B.w2f, T.w2b, B.tA2, BLL, 0);
  k_pmask<<<(BLL * 64 + 255) / 256, 256, 0, s>>>(emb_p, nb_pad, B.mL2, B.tB2, BLL, L);
  k_lin<64><<<BLL / 64, 128, 0, s>>>(B.tB2, nullptr, 64, B.w2f, T.w2b, B.tB2, BLL, 0);
  k_lin<128><<<BLL / 64, 128, 0, s>>>(B.tA2, B.tB2, 64, B.a2f, T.a2b, B.tB2, BLL, 2);
  k_dot<<<(BLL + 7) / 8, 256, 0, s>>>(B.tB2, 64, T.a2vw, T.a2vb, B.s2, BLL);
  k_attwsum<<<(BL + 7) / 8, 256, 0, s>>>(B.s2, B.mL2, B.xL2, B.sum2, BL, L);
  k_lin<64><<<BL / 64, 128, 0, s>>>(B.sum2, nullptr, 64, B.ag2f, T.ag2b, B.hL2, BL, 1);
  if (is_user) {
    k_gru<<<BL / 128, 256, 0, s>>>(B.xL2, 30LL * 30 * 64, 30LL * 64, 64LL, T.h0,
                                   B.wihf, B.whhf, B.bihp, B.bhhp, B.seqf, B.hS2, BL, 30, 30);
    k_mul<<<(BL * 64 + 255) / 256, 256, 0, s>>>(B.hL2, B.hS2, B.hI2, BL * 64);
  } else {
    k_mulb<<<(BL * 64 + 255) / 256, 256, 0, s>>>(B.hL2, B.hS1, B.hI2, BL, L);
  }
  // attention 3 (over neighbors)
  k_lin<64><<<BL / 64, 128, 0, s>>>(B.hI2, nullptr, 64, B.w3f, T.w3b, B.tA2, BL, 0);
  k_gatherE<<<(BL * 64 + 255) / 256, 256, 0, s>>>(emb_p, nb_pad, B.tB1, BL);
  k_lin<64><<<BL / 64, 128, 0, s>>>(B.tB1, nullptr, 64, B.w3f, T.w3b, B.tB1, BL, 0);
  k_lin<128><<<BL / 64, 128, 0, s>>>(B.tA2, B.tB1, 64, B.a3f, T.a3b, B.tB1, BL, 2);
  k_dot<<<(BL + 7) / 8, 256, 0, s>>>(B.tB1, 64, T.a3vw, T.a3vb, B.s3, BL);
  k_maskint<<<(BL + 255) / 256, 256, 0, s>>>(nb_pad, B.mSU, BL);
  k_attwsum<<<(BN + 7) / 8, 256, 0, s>>>(B.s3, B.mSU, B.hI2, B.sum3, BN, L);
  k_lin<64><<<BN / 64, 128, 0, s>>>(B.sum3, nullptr, 64, B.ag3f, T.ag3b, B.hN, BN, 1);

  // combine
  k_lin<128><<<BN / 64, 128, 0, s>>>(B.hI1, B.hN, 128, B.c1f, T.c1b, B.c1, BN, 1);
  k_lin<128><<<BN / 64, 128, 0, s>>>(B.c1, nullptr, 64, B.c2f, T.c2b, towerOut, BN, 1);
}

extern "C" void kernel_launch(void* const* d_in, const int* in_sizes, int n_in,
                              void* d_out, int out_size, void* d_ws, size_t ws_size,
                              hipStream_t stream) {
  (void)in_sizes; (void)n_in; (void)out_size; (void)ws_size;
  const int BN = 256, L = 30, BL = BN * L, BLL = BL * L;

  const int* uids            = (const int*)d_in[0];
  const int* iids            = (const int*)d_in[1];
  const int* u_item_pad      = (const int*)d_in[2];
  const int* u_user_pad      = (const int*)d_in[3];
  const int* u_user_item_pad = (const int*)d_in[4];
  const int* i_user_pad      = (const int*)d_in[5];
  const int* i_item_pad      = (const int*)d_in[6];
  const int* i_item_user_pad = (const int*)d_in[7];

  int c = 8;
  const float* user_emb = (const float*)d_in[c++];
  const float* item_emb = (const float*)d_in[c++];
  const float* rate_emb = (const float*)d_in[c++];
  TowerP TU = parse_tower(d_in, c);
  TowerP TI = parse_tower(d_in, c);
  const float* p1w = (const float*)d_in[c++];
  const float* p1b = (const float*)d_in[c++];
  const float* p2w = (const float*)d_in[c++];
  const float* p2b = (const float*)d_in[c++];

  char* base = (char*)d_ws; size_t off = 0;
  auto alloc = [&](size_t bytes) -> void* {
    void* p = base + off; off = (off + bytes + 255) & ~(size_t)255; return p;
  };

  Bufs B;
  B.xL1  = (float*)alloc((size_t)BL * 64 * 4);   B.mL1 = (float*)alloc((size_t)BL * 4);
  B.tA1  = (float*)alloc((size_t)BL * 64 * 4);   B.tB1 = (float*)alloc((size_t)BL * 64 * 4);
  B.s1   = (float*)alloc((size_t)BL * 4);
  B.sum1 = (float*)alloc((size_t)BN * 64 * 4);   B.hL1 = (float*)alloc((size_t)BN * 64 * 4);
  B.hS1  = (float*)alloc((size_t)BN * 64 * 4);   B.hI1 = (float*)alloc((size_t)BN * 64 * 4);
  B.xL2  = (float*)alloc((size_t)BLL * 64 * 4);  B.mL2 = (float*)alloc((size_t)BLL * 4);
  B.tA2  = (float*)alloc((size_t)BLL * 64 * 4);
  B.tB2  = (float*)alloc((size_t)BLL * 64 * 4);
  B.seqf = (_Float16*)B.tB2;                     // disjoint lifetime alias
  B.s2   = (float*)alloc((size_t)BLL * 4);
  B.sum2 = (float*)alloc((size_t)BL * 64 * 4);   B.hL2 = (float*)alloc((size_t)BL * 64 * 4);
  B.hS2  = (float*)alloc((size_t)BL * 64 * 4);   B.hI2 = (float*)alloc((size_t)BL * 64 * 4);
  B.mSU  = (float*)alloc((size_t)BL * 4);        B.s3  = (float*)alloc((size_t)BL * 4);
  B.sum3 = (float*)alloc((size_t)BN * 64 * 4);   B.hN  = (float*)alloc((size_t)BN * 64 * 4);
  B.c1   = (float*)alloc((size_t)BN * 128 * 4);
  B.w1f  = (_Float16*)alloc(4096 * 2);  B.w2f  = (_Float16*)alloc(4096 * 2);
  B.w3f  = (_Float16*)alloc(4096 * 2);
  B.g1f  = (_Float16*)alloc(8192 * 2);  B.g2f  = (_Float16*)alloc(4096 * 2);
  B.a1f  = (_Float16*)alloc(8192 * 2);  B.a2f  = (_Float16*)alloc(8192 * 2);
  B.a3f  = (_Float16*)alloc(8192 * 2);
  B.ag1f = (_Float16*)alloc(4096 * 2);  B.ag2f = (_Float16*)alloc(4096 * 2);
  B.ag3f = (_Float16*)alloc(4096 * 2);
  B.c1f  = (_Float16*)alloc(16384 * 2); B.c2f  = (_Float16*)alloc(8192 * 2);
  B.wihf = (_Float16*)alloc((size_t)4 * 64 * 192 * 2);
  B.whhf = (_Float16*)alloc((size_t)4 * 64 * 192 * 2);
  B.bihp = (float*)alloc(4 * 192 * 4);  B.bhhp = (float*)alloc(4 * 192 * 4);
  _Float16* p1f = (_Float16*)alloc(8192 * 2);
  float* hU = (float*)alloc((size_t)BN * 64 * 4);
  float* hZ = (float*)alloc((size_t)BN * 64 * 4);

  run_tower(TU, true,  item_emb, user_emb, rate_emb,
            uids, u_item_pad, u_user_pad, u_user_item_pad, B, hU, stream);
  run_tower(TI, false, user_emb, item_emb, rate_emb,
            iids, i_user_pad, i_item_pad, i_item_user_pad, B, hZ, stream);

  // prediction head
  prep_w(p1w, p1f, 128, 64, stream);
  k_lin<128><<<BN / 64, 128, 0, stream>>>(hU, hZ, 64, p1f, p1b, B.c1, BN, 1);
  k_dot<<<(BN + 7) / 8, 256, 0, stream>>>(B.c1, 64, p2w, p2b, (float*)d_out, BN);
}
